// LTFF_2860448219492
// MI455X (gfx1250) — matrix operand order repaired and checked by reading, not compiled, not run on hardware
//
#include <hip/hip_runtime.h>

// ---------------------------------------------------------------------------
// LTFF forward for MI455X (gfx1250, wave32).
// Conv layers: implicit-GEMM on v_wmma_f32_16x16x32_bf16 (f32 accumulate),
// LDS-staged B tiles shared across a 4-wave block, kernel-position-major
// weight layout so im2col decomposition is shift/mask only, branchless
// vectorized LDS fill (ds_store_b128 per thread per K-step).
// Irregular gather/softmax/fold stages are bandwidth-bound VALU kernels.
// ---------------------------------------------------------------------------

typedef __bf16 bf16_t;
typedef __attribute__((ext_vector_type(16))) __bf16 v16bf;
typedef __attribute__((ext_vector_type(8)))  __bf16 v8bf;
typedef __attribute__((ext_vector_type(8)))  float  v8f;

#define Nn   2
#define Tt   3
#define Bb   6          // Nn*Tt
#define Cc   128
#define Hh   192
#define HSs  48
#define HW_S 2304       // 48*48
#define HW_F 36864      // 192*192
#define CIN  258
#define KP0  288        // CIN padded to mult of 32
#define Mm   64
#define NHh  4
#define NPp  4

// ===========================================================================
// Implicit-GEMM conv, BF16 WMMA. Block = 128 threads = 4 waves.
// Each block: 64 rows of M (4 waves x 16) by 32 cols of N (2 WMMA per wave).
// B tile (32K x 32N) staged in LDS once per K-step, shared by all 4 waves.
// Weights are [M][K] with K = r*Cin + ci (r = ky*3+kx) so ci/r = mask/shift.
// Fill: thread (q = tid>>5, nl = tid&31) owns aligned K-run [q*8, q*8+8) for
// column nl; the run never crosses an r block (Cin is 64/128), so (ky,kx) and
// the bounds test are uniform over the run -> one select + one ds_store_b128.
// ===========================================================================
template <int KS>
__global__ void wmma_conv_kernel(
    const bf16_t* __restrict__ Wp, const float* __restrict__ bias,
    const bf16_t* __restrict__ X, long xbstride,
    int Cin, int cshift, int H, int Wd, int pad, int M, int K,
    bf16_t* __restrict__ actout, long actbstride,
    float* __restrict__ f32out, long f32bstride,
    const float* __restrict__ residual, int do_leaky)
{
    __shared__ bf16_t Bs[32][40];            // [n][k], 80B rows (16B-aligned)

    const int tid  = threadIdx.x;            // 0..127
    const int lane = tid & 31;
    const int wave = tid >> 5;               // 0..3 -> M sub-tile
    const int tn   = blockIdx.x;             // 32-wide N tile
    const int tm   = blockIdx.y;             // 64-wide M tile
    const int b    = blockIdx.z;
    const int HWd  = H * Wd;
    const bf16_t* Xb = X + (long)b * xbstride;

    const int hf   = lane >> 4;
    const int l15  = lane & 15;
    const int mrow = (tm * 4 + wave) * 16 + l15;   // A: lane holds row m
    const long wbase = (long)mrow * K;

    // fill-role decomposition (hoisted out of the K loop)
    const int q  = tid >> 5;                 // K-run within step (0..3)
    const int nl = tid & 31;                 // N within tile
    const int ng = tn * 32 + nl;
    const int py = ng / Wd;
    const int px = ng - py * Wd;

    v8f acc0 = {}, acc1 = {};

    for (int kk = 0; kk < K; kk += 32) {
        if (kk + 32 < K)
            __builtin_prefetch(Wp + wbase + kk + 32, 0, 0);

        // ---- branchless cooperative im2col fill of Bs[n][k]
        {
            int kb = kk + q * 8;             // aligned 8-run start
            int cib, r;
            if (KS == 3) { cib = kb & (Cin - 1); r = kb >> cshift; }
            else         { cib = kb; r = 0; }
            int ky = r / 3, kx = r - (r / 3) * 3;
            int yy = py + ky - pad, xx = px + kx - pad;
            bool ok = (yy >= 0) && (yy < H) && (xx >= 0) && (xx < Wd);
            long soff = ok ? ((long)yy * Wd + xx) : 0;   // safe clamped address
            const bf16_t* src = Xb + (long)cib * HWd + soff;
            v8bf pack;
#pragma unroll
            for (int u = 0; u < 8; ++u) {
                bf16_t v = src[(long)u * HWd];
                pack[u] = ok ? v : (bf16_t)0.0f;
            }
            *(v8bf*)(&Bs[nl][q * 8]) = pack;             // one ds_store_b128
        }
        __syncthreads();

        // ---- A fragment (ISA 7.12.2 16-bit A 16x32): lane=M; VGPR j<4 base 0,
        //      j>=4 base 16; half-wave adds 8; pair = consecutive K.
        v16bf a;
#pragma unroll
        for (int j = 0; j < 8; ++j) {
            int k0 = kk + ((j < 4) ? 0 : 16) + 8 * hf + 2 * (j & 3);
            const bf16_t* wr = Wp + wbase + k0;
            a[2 * j]     = wr[0];
            a[2 * j + 1] = wr[1];
        }

        // ---- B fragments from LDS (same K order as A: VGPR j<4 base 0, j>=4 base 16; half-wave adds 8)
        v16bf b0, b1;
#pragma unroll
        for (int j = 0; j < 8; ++j) {
#pragma unroll
            for (int p = 0; p < 2; ++p) {
                int ko = ((j < 4) ? 0 : 16) + 8 * hf + 2 * (j & 3) + p;
                b0[2 * j + p] = Bs[l15][ko];
                b1[2 * j + p] = Bs[16 + l15][ko];
            }
        }

        acc0 = __builtin_amdgcn_wmma_f32_16x16x32_bf16(
            false, a, false, b0, (short)0, acc0, false, false);
        acc1 = __builtin_amdgcn_wmma_f32_16x16x32_bf16(
            false, a, false, b1, (short)0, acc1, false, false);
        __syncthreads();
    }

    // ---- D: element (m = vgpr + 8*hf, n = lane&15) per 16x16 sub-tile
#pragma unroll
    for (int v = 0; v < 8; ++v) {
        int m = (tm * 4 + wave) * 16 + v + 8 * hf;
        float bval = bias[m];
#pragma unroll
        for (int tile = 0; tile < 2; ++tile) {
            int n = tn * 32 + tile * 16 + l15;
            float val = (tile == 0 ? acc0[v] : acc1[v]) + bval;
            if (do_leaky) val = (val >= 0.f) ? val : 0.1f * val;
            long o = (long)m * HWd + n;
            if (actout) actout[(long)b * actbstride + o] = (bf16_t)val;
            if (f32out) {
                float r = residual ? residual[(long)b * f32bstride + o] : 0.f;
                f32out[(long)b * f32bstride + o] = val + r;
            }
        }
    }
}

// ===========================================================================
// Weight packing to bf16.  3x3 weights reordered to K = r*Cin + ci.
// ===========================================================================
__global__ void pack_weights_kernel(
    const float* __restrict__ so_w0, const float* __restrict__ aw_w0,
    const float* __restrict__ so_w1, const float* __restrict__ so_w2,
    const float* __restrict__ aw_w1, const float* __restrict__ aw_w2,
    const float* __restrict__ fus_w,
    const float* __restrict__ so_b0, const float* __restrict__ aw_b0,
    bf16_t* __restrict__ W0p, bf16_t* __restrict__ W1s, bf16_t* __restrict__ W2s,
    bf16_t* __restrict__ W1a, bf16_t* __restrict__ W2a, bf16_t* __restrict__ Wf,
    float* __restrict__ bias0)
{
    const int E0 = 128 * KP0;
    const int E1 = Mm * 576;        // 3x3 64->64
    const int E5 = Cc * 1152;       // fusion 3x3 128->128
    long i = (long)blockIdx.x * blockDim.x + threadIdx.x;

    if (i < E0) {   // stacked so|aw 1x1, K-padded, plain [o][ci]
        int o = (int)(i / KP0), k = (int)(i % KP0);
        float v = 0.f;
        if (k < CIN) v = (o < Mm) ? so_w0[o * CIN + k] : aw_w0[(o - Mm) * CIN + k];
        W0p[i] = (bf16_t)v; return;
    }
    i -= E0;
    // 3x3 64->64: dst[o][r*64+ci] = src[o*576 + ci*9 + r]
    if (i < 4 * E1) {
        int which = (int)(i / E1);
        int j = (int)(i % E1);
        int o = j / 576, rem = j % 576;
        int r = rem >> 6, ci = rem & 63;
        const float* src = (which == 0) ? so_w1 : (which == 1) ? so_w2
                         : (which == 2) ? aw_w1 : aw_w2;
        bf16_t* dst = (which == 0) ? W1s : (which == 1) ? W2s
                    : (which == 2) ? W1a : W2a;
        dst[j] = (bf16_t)src[o * 576 + ci * 9 + r];
        return;
    }
    i -= 4 * E1;
    // fusion 3x3 128->128: dst[o][r*128+ci] = src[o*1152 + ci*9 + r]
    if (i < E5) {
        int o = (int)(i / 1152), rem = (int)(i % 1152);
        int r = rem >> 7, ci = rem & 127;
        Wf[i] = (bf16_t)fus_w[o * 1152 + ci * 9 + r];
        return;
    }
    i -= E5;
    if (i < 128) bias0[i] = (i < Mm) ? so_b0[i] : aw_b0[i - Mm];
}

// ===========================================================================
// Warp (grid1) nearest-sample index: ix = round(flow+ref), valid in [0,47]
// ===========================================================================
__global__ void widx_kernel(const float* __restrict__ flow, int* __restrict__ widx)
{
    int i = blockIdx.x * blockDim.x + threadIdx.x;
    if (i >= Bb * HW_S) return;
    int b = i / HW_S, pos = i % HW_S;
    int y = pos / HSs, x = pos % HSs;
    float fx = roundf(flow[(b * 2 + 0) * HW_S + pos] + (float)x);
    float fy = roundf(flow[(b * 2 + 1) * HW_S + pos] + (float)y);
    bool ok = (fx >= 0.f) && (fx <= 47.f) && (fy >= 0.f) && (fy <= 47.f);
    widx[i] = ok ? ((int)fy * HSs + (int)fx) : -1;
}

// avgpool(fold(sp)) == mean over the 16 S*S sub-channels; writes extra rows 0..127
__global__ void pool_kernel(const float* __restrict__ sp,
                            float* __restrict__ pool, bf16_t* __restrict__ extra)
{
    long i = (long)blockIdx.x * blockDim.x + threadIdx.x;
    if (i >= (long)Bb * Cc * HW_S) return;
    int b = (int)(i / (Cc * HW_S));
    int r = (int)(i % (Cc * HW_S));
    int cc = r / HW_S, pos = r % HW_S;
    float s = 0.f;
#pragma unroll
    for (int j = 0; j < 16; ++j)
        s += sp[((long)b * 2048 + cc * 16 + j) * HW_S + pos];
    s *= (1.0f / 16.0f);
    pool[i] = s;
    extra[((long)b * KP0 + cc) * HW_S + pos] = (bf16_t)s;
}

// extra rows 128..287: warped pool gather, flow, and K-pad zeros
__global__ void extra2_kernel(const float* __restrict__ pool,
                              const int* __restrict__ widx,
                              const float* __restrict__ flow,
                              bf16_t* __restrict__ extra)
{
    long i = (long)blockIdx.x * blockDim.x + threadIdx.x;
    if (i >= (long)Bb * 160 * HW_S) return;
    int b = (int)(i / (160 * HW_S));
    int r = (int)(i % (160 * HW_S));
    int ch = 128 + r / HW_S, pos = r % HW_S;
    float v = 0.f;
    if (ch < 256) {
        int wi = widx[b * HW_S + pos];
        if (wi >= 0) v = pool[((long)b * Cc + (ch - 128)) * HW_S + wi];
    } else if (ch < 258) {
        v = flow[(b * 2 + (ch - 256)) * HW_S + pos];
    }
    extra[((long)b * KP0 + ch) * HW_S + pos] = (bf16_t)v;
}

// Tiny 1x1 heads (K=64): offsets (M=32) + raw attention (M=16). VALU is fine.
__global__ void head_kernel(const bf16_t* __restrict__ a3s, const bf16_t* __restrict__ a3a,
                            const float* __restrict__ so_w3, const float* __restrict__ so_b3,
                            const float* __restrict__ aw_w3, const float* __restrict__ aw_b3,
                            float* __restrict__ offs, float* __restrict__ attn_raw)
{
    long i = (long)blockIdx.x * blockDim.x + threadIdx.x;
    if (i >= (long)Bb * 48 * HW_S) return;
    int b = (int)(i / (48 * HW_S));
    int r = (int)(i % (48 * HW_S));
    int o = r / HW_S, pos = r % HW_S;
    if (o < 32) {
        float acc = so_b3[o];
        const bf16_t* xb = a3s + (long)b * Mm * HW_S;
#pragma unroll 8
        for (int ci = 0; ci < Mm; ++ci)
            acc += so_w3[o * Mm + ci] * (float)xb[ci * HW_S + pos];
        offs[((long)b * 32 + o) * HW_S + pos] = acc;
    } else {
        int o2 = o - 32;
        float acc = aw_b3[o2];
        const bf16_t* xb = a3a + (long)b * Mm * HW_S;
#pragma unroll 8
        for (int ci = 0; ci < Mm; ++ci)
            acc += aw_w3[o2 * Mm + ci] * (float)xb[ci * HW_S + pos];
        attn_raw[((long)b * 16 + o2) * HW_S + pos] = acc;
    }
}

// softmax over (t, NP) = 12 values per (n, head, pos)
__global__ void softmax_kernel(const float* __restrict__ attn_raw, float* __restrict__ attn_sm)
{
    int i = blockIdx.x * blockDim.x + threadIdx.x;
    if (i >= Nn * NHh * HW_S) return;
    int nn = i / (NHh * HW_S);
    int r = i % (NHh * HW_S);
    int head = r / HW_S, pos = r % HW_S;
    float v[12], mx = -3.4e38f;
#pragma unroll
    for (int t = 0; t < Tt; ++t)
#pragma unroll
        for (int p = 0; p < NPp; ++p) {
            int j = t * NPp + p;
            v[j] = attn_raw[(((long)(nn * Tt + t) * 16) + head * NPp + p) * HW_S + pos];
            mx = fmaxf(mx, v[j]);
        }
    float s = 0.f;
#pragma unroll
    for (int j = 0; j < 12; ++j) { v[j] = expf(v[j] - mx); s += v[j]; }
    float inv = 1.0f / s;
#pragma unroll
    for (int j = 0; j < 12; ++j) attn_sm[(long)i * 12 + j] = v[j] * inv;
}

// nearest-sample index for deformable points: round(og * 47/48), valid in [0,47]
__global__ void sampidx_kernel(const float* __restrict__ flow, const float* __restrict__ offs,
                               int* __restrict__ samp)
{
    int i = blockIdx.x * blockDim.x + threadIdx.x;
    if (i >= Nn * Tt * NHh * NPp * HW_S) return;
    int pos = i % HW_S; int r = i / HW_S;
    int p = r % NPp; r /= NPp;
    int head = r % NHh; r /= NHh;
    int t = r % Tt; int nn = r / Tt;
    int b = nn * Tt + t;
    int y = pos / HSs, x = pos % HSs;
    int hp = head * NPp + p;
    float ogx = flow[(b * 2 + 0) * HW_S + pos] + (float)x
              + offs[((long)b * 32 + hp * 2 + 0) * HW_S + pos];
    float ogy = flow[(b * 2 + 1) * HW_S + pos] + (float)y
              + offs[((long)b * 32 + hp * 2 + 1) * HW_S + pos];
    float fx = roundf(ogx * (47.0f / 48.0f));
    float fy = roundf(ogy * (47.0f / 48.0f));
    bool ok = (fx >= 0.f) && (fx <= 47.f) && (fy >= 0.f) && (fy <= 47.f);
    samp[i] = ok ? ((int)fy * HSs + (int)fx) : -1;
}

// Gather 12 (t,p) samples per channel, weight by softmax attn, fold directly
// into the hidden feature map (bf16 for the fusion WMMA conv).
__global__ void accum_kernel(const float* __restrict__ sp,
                             const float* __restrict__ attn_sm,
                             const int* __restrict__ samp,
                             bf16_t* __restrict__ hidden)
{
    __shared__ float sw[12];
    __shared__ int sidx[12];
    int g = blockIdx.x;                  // (n, head, pos)
    int pos = g % HW_S; int r = g / HW_S;
    int head = r % NHh; int nn = r / NHh;
    int tid = threadIdx.x;
    if (tid < 12) {
        int t = tid / NPp, p = tid % NPp;
        sw[tid] = attn_sm[(long)g * 12 + tid];
        sidx[tid] = samp[((((nn * Tt + t) * NHh + head) * NPp + p) * (long)HW_S) + pos];
    }
    __syncthreads();
    int ys = pos / HSs, xs = pos % HSs;
#pragma unroll
    for (int rep = 0; rep < 2; ++rep) {
        int k = tid + rep * 256;         // 512 channels per (n,head,pos)
        float acc = 0.f;
#pragma unroll
        for (int j = 0; j < 12; ++j) {
            int si = sidx[j];
            if (si >= 0) {
                int b = nn * Tt + (j >> 2);
                acc += sw[j] * sp[((long)b * 2048 + head * 512 + k) * HW_S + si];
            }
        }
        int cc2 = k >> 4, s = k & 15;
        int y = ys * 4 + (s >> 2), x = xs * 4 + (s & 3);
        int ch = head * 32 + cc2;
        hidden[(((long)nn * Cc + ch) * Hh + y) * Hh + x] = (bf16_t)acc;
    }
}

// ===========================================================================
// Host launch
// ===========================================================================
static inline size_t al256(size_t x) { return (x + 255) & ~(size_t)255; }

extern "C" void kernel_launch(void* const* d_in, const int* in_sizes, int n_in,
                              void* d_out, int out_size, void* d_ws, size_t ws_size,
                              hipStream_t stream) {
    const float* anchor = (const float*)d_in[1];
    const float* sp     = (const float*)d_in[2];
    const float* flow   = (const float*)d_in[3];
    const float* so_w0 = (const float*)d_in[4],  *so_b0 = (const float*)d_in[5];
    const float* so_w1 = (const float*)d_in[6],  *so_b1 = (const float*)d_in[7];
    const float* so_w2 = (const float*)d_in[8],  *so_b2 = (const float*)d_in[9];
    const float* so_w3 = (const float*)d_in[10], *so_b3 = (const float*)d_in[11];
    const float* aw_w0 = (const float*)d_in[12], *aw_b0 = (const float*)d_in[13];
    const float* aw_w1 = (const float*)d_in[14], *aw_b1 = (const float*)d_in[15];
    const float* aw_w2 = (const float*)d_in[16], *aw_b2 = (const float*)d_in[17];
    const float* aw_w3 = (const float*)d_in[18], *aw_b3 = (const float*)d_in[19];
    const float* fus_w = (const float*)d_in[20], *fus_b = (const float*)d_in[21];
    float* out = (float*)d_out;

    // ---- workspace carve-up
    char* base = (char*)d_ws; size_t off = 0;
    auto take = [&](size_t bytes) { char* p = base + off; off = al256(off + bytes); return p; };
    float*  pool    = (float*) take((size_t)Bb * Cc * HW_S * 4);
    int*    widx    = (int*)   take((size_t)Bb * HW_S * 4);
    bf16_t* extra   = (bf16_t*)take((size_t)Bb * KP0 * HW_S * 2);
    bf16_t* W0p     = (bf16_t*)take((size_t)128 * KP0 * 2);
    bf16_t* W1s     = (bf16_t*)take((size_t)Mm * 576 * 2);
    bf16_t* W2s     = (bf16_t*)take((size_t)Mm * 576 * 2);
    bf16_t* W1a     = (bf16_t*)take((size_t)Mm * 576 * 2);
    bf16_t* W2a     = (bf16_t*)take((size_t)Mm * 576 * 2);
    bf16_t* Wf      = (bf16_t*)take((size_t)Cc * 1152 * 2);
    float*  bias0   = (float*) take(128 * 4);
    bf16_t* act1    = (bf16_t*)take((size_t)Bb * 128 * HW_S * 2);   // stacked so|aw
    bf16_t* a2s     = (bf16_t*)take((size_t)Bb * Mm * HW_S * 2);
    bf16_t* a2a     = (bf16_t*)take((size_t)Bb * Mm * HW_S * 2);
    bf16_t* a3s     = (bf16_t*)take((size_t)Bb * Mm * HW_S * 2);
    bf16_t* a3a     = (bf16_t*)take((size_t)Bb * Mm * HW_S * 2);
    float*  offs    = (float*) take((size_t)Bb * 32 * HW_S * 4);
    float*  attn_rw = (float*) take((size_t)Bb * 16 * HW_S * 4);
    float*  attn_sm = (float*) take((size_t)Nn * NHh * HW_S * 12 * 4);
    int*    samp    = (int*)   take((size_t)Nn * Tt * NHh * NPp * HW_S * 4);
    bf16_t* hidden  = (bf16_t*)take((size_t)Nn * Cc * HW_F * 2);
    (void)ws_size; (void)in_sizes; (void)n_in; (void)out_size;

    // ---- 1. pack weights to bf16 (3x3 reordered to r-major K)
    {
        long tot = (long)128 * KP0 + 4L * Mm * 576 + (long)Cc * 1152 + 128;
        pack_weights_kernel<<<(tot + 255) / 256, 256, 0, stream>>>(
            so_w0, aw_w0, so_w1, so_w2, aw_w1, aw_w2, fus_w, so_b0, aw_b0,
            W0p, W1s, W2s, W1a, W2a, Wf, bias0);
    }
    // ---- 2. warp indices, pooled features, extra tensor (K-padded bf16)
    widx_kernel<<<(Bb * HW_S + 255) / 256, 256, 0, stream>>>(flow, widx);
    pool_kernel<<<((long)Bb * Cc * HW_S + 255) / 256, 256, 0, stream>>>(sp, pool, extra);
    extra2_kernel<<<((long)Bb * 160 * HW_S + 255) / 256, 256, 0, stream>>>(pool, widx, flow, extra);

    // ---- 3. conv0 1x1 (stacked so|aw): M=128, K=288, N=2304, WMMA
    wmma_conv_kernel<1><<<dim3(HW_S / 32, 128 / 64, Bb), 128, 0, stream>>>(
        W0p, bias0, extra, (long)KP0 * HW_S, KP0, 0, HSs, HSs, 0, 128, KP0,
        act1, (long)128 * HW_S, nullptr, 0, nullptr, 1);

    // ---- 4. 3x3 convs 64->64 (so/aw, layer1/layer2), WMMA implicit GEMM
    wmma_conv_kernel<3><<<dim3(HW_S / 32, 1, Bb), 128, 0, stream>>>(
        W1s, so_b1, act1, (long)128 * HW_S, Mm, 6, HSs, HSs, 1, Mm, 576,
        a2s, (long)Mm * HW_S, nullptr, 0, nullptr, 1);
    wmma_conv_kernel<3><<<dim3(HW_S / 32, 1, Bb), 128, 0, stream>>>(
        W1a, aw_b1, act1 + (long)Mm * HW_S, (long)128 * HW_S, Mm, 6, HSs, HSs, 1, Mm, 576,
        a2a, (long)Mm * HW_S, nullptr, 0, nullptr, 1);
    wmma_conv_kernel<3><<<dim3(HW_S / 32, 1, Bb), 128, 0, stream>>>(
        W2s, so_b2, a2s, (long)Mm * HW_S, Mm, 6, HSs, HSs, 1, Mm, 576,
        a3s, (long)Mm * HW_S, nullptr, 0, nullptr, 1);
    wmma_conv_kernel<3><<<dim3(HW_S / 32, 1, Bb), 128, 0, stream>>>(
        W2a, aw_b2, a2a, (long)Mm * HW_S, Mm, 6, HSs, HSs, 1, Mm, 576,
        a3a, (long)Mm * HW_S, nullptr, 0, nullptr, 1);

    // ---- 5. 1x1 heads -> offsets + raw attention
    head_kernel<<<((long)Bb * 48 * HW_S + 255) / 256, 256, 0, stream>>>(
        a3s, a3a, so_w3, so_b3, aw_w3, aw_b3, offs, attn_rw);

    // ---- 6. softmax over t*NP, sample indices
    softmax_kernel<<<(Nn * NHh * HW_S + 255) / 256, 256, 0, stream>>>(attn_rw, attn_sm);
    sampidx_kernel<<<(Nn * Tt * NHh * NPp * HW_S + 255) / 256, 256, 0, stream>>>(flow, offs, samp);

    // ---- 7. gather + attention-weighted sum + fold -> hidden (bf16)
    accum_kernel<<<Nn * NHh * HW_S, 256, 0, stream>>>(sp, attn_sm, samp, hidden);

    // ---- 8. fusion 3x3 conv 128->128 @192x192, fused bias + anchor residual
    wmma_conv_kernel<3><<<dim3(HW_F / 32, 128 / 64, Nn), 128, 0, stream>>>(
        Wf, fus_b, hidden, (long)Cc * HW_F, Cc, 7, Hh, Hh, 1, Cc, 1152,
        nullptr, 0, out, (long)Cc * HW_F, anchor, 0);
}
